// GemmaAttention_73993696575844
// MI455X (gfx1250) — matrix operand order repaired and checked by reading, not compiled, not run on hardware
//
#include <hip/hip_runtime.h>

// ---------------- problem constants ----------------
#define Bsz   2
#define Sln   2048
#define Hdim  3072
#define NH    16
#define NKV   16
#define HD    256
#define QSZ   (NH * HD)            // 4096
#define KVSZ  (NKV * HD)           // 4096
#define QKVW  (QSZ + 2 * KVSZ)     // 12288
#define ROWS  (Bsz * Sln)          // 4096

typedef __bf16 bf16x16 __attribute__((ext_vector_type(16)));
typedef float  f32x8   __attribute__((ext_vector_type(8)));
typedef unsigned int v4u __attribute__((ext_vector_type(4)));
typedef unsigned int v8u __attribute__((ext_vector_type(8)));

union FragU { bf16x16 bf; uint4 u4[2]; };
union AccU  { f32x8 v; float f[8]; };

typedef __attribute__((address_space(3))) unsigned char lds_byte;

__device__ __forceinline__ unsigned short f2bf(float x) {
    __bf16 h = (__bf16)x;                       // hardware RNE convert
    return __builtin_bit_cast(unsigned short, h);
}
__device__ __forceinline__ float bf2f(unsigned short h) {
    return __uint_as_float(((unsigned)h) << 16);
}

__device__ __forceinline__ f32x8 wmma_bf16(bf16x16 a, bf16x16 b, f32x8 c) {
    // (neg_a, A, neg_b, B, c_mod, C, reuse_a, reuse_b)
    return __builtin_amdgcn_wmma_f32_16x16x32_bf16(false, a, false, b, (short)0, c,
                                                   false, false);
}

__device__ __forceinline__ unsigned lds_off_of(const void* l) {
    return (unsigned)(unsigned long long)(lds_byte*)l;
}

// Async global -> LDS copy of 16 bytes (CDNA5, tracked by ASYNCcnt).
__device__ __forceinline__ void async_load_b128(const void* g, void* l) {
    unsigned lds_off = lds_off_of(l);
    unsigned long long ga = (unsigned long long)g;
    asm volatile("global_load_async_to_lds_b128 %0, %1, off"
                 :: "v"(lds_off), "v"(ga) : "memory");
}
__device__ __forceinline__ void wait_async0() {
    asm volatile("s_wait_asynccnt 0x0" ::: "memory");
}

// Tensor Data Mover: DMA a 2D tile of bf16 (tile_w x tile_h elements, row
// stride row_stride elements) from global memory into LDS (packed row-major).
// D# layout per CDNA5 ISA (2D: groups 2/3 NULL).
__device__ __forceinline__ void tdm_load_2d_bf16(
    unsigned lds_off, const void* gptr,
    unsigned tile_w, unsigned tile_h,
    unsigned tensor_w, unsigned tensor_h, unsigned row_stride)
{
    unsigned long long ga = (unsigned long long)gptr;
    v4u g0;
    g0[0] = 1u;                                           // count=1 (valid), user D#
    g0[1] = lds_off;                                      // lds_addr (bytes)
    g0[2] = (unsigned)(ga & 0xFFFFFFFFu);                 // global_addr[31:0]
    g0[3] = (unsigned)((ga >> 32) & 0x1FFFFFFu) | (2u << 30);  // addr[56:32] | type=2
    v8u g1;
    g1[0] = 1u << 16;                                     // wg_mask=0, data_size=1 (2B)
    g1[1] = (tensor_w & 0xFFFFu) << 16;                   // tensor_dim0[15:0]
    g1[2] = (tensor_w >> 16) | ((tensor_h & 0xFFFFu) << 16);   // dim0[31:16] | dim1[15:0]
    g1[3] = (tensor_h >> 16) | (tile_w << 16);            // dim1[31:16] | tile_dim0
    g1[4] = tile_h;                                       // tile_dim1, tile_dim2=0
    g1[5] = row_stride;                                   // tensor_dim0_stride[31:0]
    g1[6] = 0u;                                           // stride0[47:32], dim1_stride lo
    g1[7] = 0u;
    asm volatile("tensor_load_to_lds %0, %1" :: "s"(g0), "s"(g1) : "memory");
}
__device__ __forceinline__ void wait_tensor0() {
    __builtin_amdgcn_s_wait_tensorcnt(0);
}

// ---------------------------------------------------------------------------
// Tiled GEMM: C[M,N] = A[M,K] * B[K,N].  Tile 128x128x32, 256 threads (8
// waves); wave w computes a 16x128 strip via 8 v_wmma_f32_16x16x32_bf16
// accumulators.  Global float4 loads for the next K-tile are register-
// prefetched so HBM latency overlaps the WMMA block.  A is fp32 (converted)
// or bf16 (async-copied to LDS), B is fp32 (converted + transposed to LDS).
// ---------------------------------------------------------------------------
template <int A_F32, int C_F32>
__global__ __launch_bounds__(256) void gemm_bf16_wmma(
    const void* __restrict__ Ap, const float* __restrict__ Bp, void* __restrict__ Cp,
    int K, int lda, int ldb, int ldc)
{
    __shared__ alignas(16) unsigned short As[128 * 32];   // [m][k]
    __shared__ alignas(16) unsigned short Bs[128 * 32];   // [n][k]  (B transposed)

    const int t    = threadIdx.x;
    const int w    = t >> 5;
    const int lane = t & 31;
    const int lr   = lane & 15;
    const int hi   = lane >> 4;
    const int m0   = blockIdx.y * 128;
    const int n0   = blockIdx.x * 128;

    AccU acc[8];
    #pragma unroll
    for (int nt = 0; nt < 8; ++nt)
        for (int j = 0; j < 8; ++j) acc[nt].f[j] = 0.0f;

    float4 aReg[4], bReg[4];

    auto load_regs = [&](int k0) {
        if (A_F32) {
            const float* A = (const float*)Ap;
            #pragma unroll
            for (int i = 0; i < 4; ++i) {               // 1024 float4 chunks
                int idx = t + 256 * i;
                int r = idx >> 3, c4 = (idx & 7) * 4;
                aReg[i] = *(const float4*)(A + (size_t)(m0 + r) * lda + k0 + c4);
            }
        }
        #pragma unroll
        for (int i = 0; i < 4; ++i) {                   // B tile 32x128
            int idx = t + 256 * i;
            int r = idx >> 5, c4 = (idx & 31) * 4;
            bReg[i] = *(const float4*)(Bp + (size_t)(k0 + r) * ldb + n0 + c4);
        }
    };
    auto store_lds = [&](int k0) {
        if (A_F32) {
            #pragma unroll
            for (int i = 0; i < 4; ++i) {
                int idx = t + 256 * i;
                int r = idx >> 3, c4 = (idx & 7) * 4;
                uint2 p;
                p.x = (unsigned)f2bf(aReg[i].x) | ((unsigned)f2bf(aReg[i].y) << 16);
                p.y = (unsigned)f2bf(aReg[i].z) | ((unsigned)f2bf(aReg[i].w) << 16);
                *(uint2*)&As[r * 32 + c4] = p;
            }
        } else {
            const unsigned short* A = (const unsigned short*)Ap;
            #pragma unroll
            for (int i = 0; i < 2; ++i) {               // 512 b128 async chunks
                int idx = t + 256 * i;
                int r = idx >> 2, c8 = (idx & 3) * 8;
                async_load_b128(A + (size_t)(m0 + r) * lda + k0 + c8,
                                &As[r * 32 + c8]);
            }
        }
        #pragma unroll
        for (int i = 0; i < 4; ++i) {                   // transpose B into [n][k]
            int idx = t + 256 * i;
            int r = idx >> 5, c4 = (idx & 31) * 4;
            Bs[(c4 + 0) * 32 + r] = f2bf(bReg[i].x);
            Bs[(c4 + 1) * 32 + r] = f2bf(bReg[i].y);
            Bs[(c4 + 2) * 32 + r] = f2bf(bReg[i].z);
            Bs[(c4 + 3) * 32 + r] = f2bf(bReg[i].w);
        }
    };

    load_regs(0);
    for (int k0 = 0; k0 < K; k0 += 32) {
        __syncthreads();
        store_lds(k0);
        if (!A_F32) wait_async0();
        __syncthreads();
        if (k0 + 32 < K) load_regs(k0 + 32);            // prefetch next tile

        FragU a;
        a.u4[0] = *(const uint4*)&As[(w * 16 + lr) * 32 + hi * 8];
        a.u4[1] = *(const uint4*)&As[(w * 16 + lr) * 32 + 16 + hi * 8];
        #pragma unroll
        for (int nt = 0; nt < 8; ++nt) {
            FragU b;
            b.u4[0] = *(const uint4*)&Bs[(nt * 16 + lr) * 32 + hi * 8];
            b.u4[1] = *(const uint4*)&Bs[(nt * 16 + lr) * 32 + 16 + hi * 8];
            acc[nt].v = wmma_bf16(a.bf, b.bf, acc[nt].v);
        }
    }

    #pragma unroll
    for (int nt = 0; nt < 8; ++nt)
        for (int j = 0; j < 8; ++j) {
            int row = m0 + w * 16 + j + 8 * hi;
            int col = n0 + nt * 16 + lr;
            if (C_F32)
                ((float*)Cp)[(size_t)row * ldc + col] = acc[nt].f[j];
            else
                ((unsigned short*)Cp)[(size_t)row * ldc + col] = f2bf(acc[nt].f[j]);
        }
}

// ---------------------------------------------------------------------------
// NeoX RoPE applied in place to the q and k sections of the bf16 qkv buffer.
// ---------------------------------------------------------------------------
__global__ __launch_bounds__(128) void rope_kernel(unsigned short* __restrict__ qkv,
                                                   const int* __restrict__ positions)
{
    const int rh   = blockIdx.x;
    const int row  = rh >> 5;          // 0..ROWS-1
    const int head = rh & 31;          // 0..15 q heads, 16..31 k heads
    const int j    = threadIdx.x;      // pair index 0..127

    const int pos = positions[row];
    const float inv_freq = __expf(-(float)(2 * j) * (9.210340371976184f / (float)HD));
    const float ang = (float)pos * inv_freq;
    float s, c;
    __sincosf(ang, &s, &c);

    const size_t base = (size_t)row * QKVW +
                        (head < 16 ? head * HD : QSZ + (head - 16) * HD);
    const float x1 = bf2f(qkv[base + j]);
    const float x2 = bf2f(qkv[base + j + HD / 2]);
    qkv[base + j]           = f2bf(x1 * c - x2 * s);
    qkv[base + j + HD / 2]  = f2bf(x2 * c + x1 * s);
}

// ---------------------------------------------------------------------------
// Flash attention (causal, online softmax).  One WG per (q-block of 64, head,
// batch), 128 threads = 4 waves; wave w owns q rows [w*16, w*16+16) entirely so
// softmax statistics are wave-local.  The K tile (32x256 bf16, strided rows) is
// DMA'd by the Tensor Data Mover from wave 0 (TENSORcnt); V is transposed via
// DS stores.  LDS: K[32][256], V^T[256][32], P[64][32]  (36 KB).
// ---------------------------------------------------------------------------
__global__ __launch_bounds__(128) void flash_attn_kernel(
    const unsigned short* __restrict__ qkv, unsigned short* __restrict__ attn)
{
    __shared__ alignas(16) unsigned short Ks[32 * 256];  // [k][d]
    __shared__ alignas(16) unsigned short Vt[256 * 32];  // [d][k]
    __shared__ alignas(16) unsigned short Ps[64 * 32];   // [q][k], wave-private rows

    const int t    = threadIdx.x;
    const int w    = t >> 5;
    const int lane = t & 31;
    const int lr   = lane & 15;
    const int hi   = lane >> 4;
    const int qb   = blockIdx.x;
    const int h    = blockIdx.y;
    const int b    = blockIdx.z;
    const int q0   = qb * 64;

    const size_t bbase = (size_t)b * Sln * QKVW;

    // Preload Q fragments for this wave's 16 rows (bf16, 8 K-steps of 32 over HD)
    FragU qf[8];
    {
        const unsigned short* qrow =
            qkv + bbase + (size_t)(q0 + w * 16 + lr) * QKVW + h * HD;
        #pragma unroll
        for (int kk = 0; kk < 8; ++kk) {
            qf[kk].u4[0] = *(const uint4*)(qrow + kk * 32 + hi * 8);
            qf[kk].u4[1] = *(const uint4*)(qrow + kk * 32 + 16 + hi * 8);
        }
    }

    AccU oacc[16];
    #pragma unroll
    for (int nt = 0; nt < 16; ++nt)
        for (int j = 0; j < 8; ++j) oacc[nt].f[j] = 0.0f;

    float mrow[8], lrow[8];
    #pragma unroll
    for (int j = 0; j < 8; ++j) { mrow[j] = -3.0e38f; lrow[j] = 0.0f; }

    const unsigned short* kbase = qkv + bbase + QSZ + h * HD;
    const unsigned short* vbase = qkv + bbase + QSZ + KVSZ + h * HD;
    const unsigned ks_off = lds_off_of(&Ks[0]);

    for (int k0 = 0; k0 <= q0 + 63; k0 += 32) {
        __syncthreads();
        // K tile: TDM 2D tile DMA (rows k0..k0+31, 256 bf16 each, stride QKVW)
        if (w == 0)
            tdm_load_2d_bf16(ks_off, kbase + (size_t)k0 * QKVW,
                             /*tile_w=*/HD, /*tile_h=*/32,
                             /*tensor_w=*/HD, /*tensor_h=*/Sln,
                             /*row_stride=*/QKVW);
        // V tile: load + transpose via DS stores.
        #pragma unroll
        for (int i = 0; i < 8; ++i) {                   // 1024 b128 chunks
            int idx = t + 128 * i;
            int r = idx >> 5, c8 = (idx & 31) * 8;
            uint4 vv = *(const uint4*)(vbase + (size_t)(k0 + r) * QKVW + c8);
            const unsigned short* vp = (const unsigned short*)&vv;
            #pragma unroll
            for (int jj = 0; jj < 8; ++jj) Vt[(c8 + jj) * 32 + r] = vp[jj];
        }
        if (w == 0) wait_tensor0();
        __syncthreads();

        // S = Q * K^T : wave's 16 q rows x 32 k cols, contract over HD=256
        AccU sacc[2];
        for (int nt = 0; nt < 2; ++nt)
            for (int j = 0; j < 8; ++j) sacc[nt].f[j] = 0.0f;
        #pragma unroll
        for (int kk = 0; kk < 8; ++kk) {
            #pragma unroll
            for (int nt = 0; nt < 2; ++nt) {
                FragU kf;
                kf.u4[0] = *(const uint4*)&Ks[(nt * 16 + lr) * 256 + kk * 32 + hi * 8];
                kf.u4[1] = *(const uint4*)&Ks[(nt * 16 + lr) * 256 + kk * 32 + 16 + hi * 8];
                sacc[nt].v = wmma_bf16(qf[kk].bf, kf.bf, sacc[nt].v);
            }
        }

        // Online softmax, wave-local per row
        float corr[8];
        #pragma unroll
        for (int j = 0; j < 8; ++j) {
            const int rowq = q0 + w * 16 + j + 8 * hi;
            float v0 = sacc[0].f[j] * 0.0625f;          // 1/sqrt(256)
            float v1 = sacc[1].f[j] * 0.0625f;
            if (k0 + lr > rowq)      v0 = -3.0e38f;     // causal mask
            if (k0 + 16 + lr > rowq) v1 = -3.0e38f;
            float tm = fmaxf(v0, v1);
            #pragma unroll
            for (int m = 1; m < 16; m <<= 1) tm = fmaxf(tm, __shfl_xor(tm, m, 32));
            const float mnew = fmaxf(mrow[j], tm);
            const float cr   = __expf(mrow[j] - mnew);
            mrow[j] = mnew;
            corr[j] = cr;
            const float p0 = __expf(v0 - mnew);
            const float p1 = __expf(v1 - mnew);
            float ps = p0 + p1;
            #pragma unroll
            for (int m = 1; m < 16; m <<= 1) ps += __shfl_xor(ps, m, 32);
            lrow[j] = lrow[j] * cr + ps;
            // stage P (bf16) into wave-private LDS rows (same-wave LDS is in-order)
            Ps[(w * 16 + j + 8 * hi) * 32 + lr]      = f2bf(p0);
            Ps[(w * 16 + j + 8 * hi) * 32 + 16 + lr] = f2bf(p1);
        }

        // Rescale running O by exp(m_old - m_new)
        #pragma unroll
        for (int nt = 0; nt < 16; ++nt)
            for (int j = 0; j < 8; ++j) oacc[nt].f[j] *= corr[j];

        // O += P * V   (contract over 32 k, 16 d-tiles of 16)
        FragU pf;
        pf.u4[0] = *(const uint4*)&Ps[(w * 16 + lr) * 32 + hi * 8];
        pf.u4[1] = *(const uint4*)&Ps[(w * 16 + lr) * 32 + 16 + hi * 8];
        #pragma unroll
        for (int nt = 0; nt < 16; ++nt) {
            FragU vf;
            vf.u4[0] = *(const uint4*)&Vt[(nt * 16 + lr) * 32 + hi * 8];
            vf.u4[1] = *(const uint4*)&Vt[(nt * 16 + lr) * 32 + 16 + hi * 8];
            oacc[nt].v = wmma_bf16(pf.bf, vf.bf, oacc[nt].v);
        }
    }

    // Epilogue: O / l  -> bf16 attn buffer [row][NH*HD]
    float rl[8];
    #pragma unroll
    for (int j = 0; j < 8; ++j) rl[j] = 1.0f / lrow[j];
    unsigned short* obase =
        attn + (size_t)(b * Sln + q0 + w * 16) * QSZ + h * HD;
    #pragma unroll
    for (int nt = 0; nt < 16; ++nt)
        for (int j = 0; j < 8; ++j)
            obase[(size_t)(j + 8 * hi) * QSZ + nt * 16 + lr] =
                f2bf(oacc[nt].f[j] * rl[j]);
}

// ---------------------------------------------------------------------------
extern "C" void kernel_launch(void* const* d_in, const int* in_sizes, int n_in,
                              void* d_out, int out_size, void* d_ws, size_t ws_size,
                              hipStream_t stream)
{
    const float* hidden    = (const float*)d_in[0];   // [B,S,H] fp32
    const int*   positions = (const int*)d_in[1];     // [B,S] i32
    const float* Wqkv      = (const float*)d_in[2];   // [H, QKVW] fp32
    const float* Wo        = (const float*)d_in[3];   // [QSZ, H] fp32
    float*       out       = (float*)d_out;           // [B,S,H] fp32

    unsigned short* qkv  = (unsigned short*)d_ws;                       // ROWS x QKVW bf16
    unsigned short* attn = (unsigned short*)((char*)d_ws +
                            (size_t)ROWS * QKVW * sizeof(unsigned short)); // ROWS x QSZ bf16

    // 1) QKV projection: [4096,3072] x [3072,12288] -> bf16 qkv
    {
        dim3 grid(QKVW / 128, ROWS / 128);
        gemm_bf16_wmma<1, 0><<<grid, 256, 0, stream>>>(
            hidden, Wqkv, qkv, Hdim, Hdim, QKVW, QKVW);
    }
    // 2) RoPE on q and k (in place)
    {
        dim3 grid(ROWS * (NH + NKV));
        rope_kernel<<<grid, 128, 0, stream>>>(qkv, positions);
    }
    // 3) Causal flash attention -> bf16 attn
    {
        dim3 grid(Sln / 64, NH, Bsz);
        flash_attn_kernel<<<grid, 128, 0, stream>>>(qkv, attn);
    }
    // 4) Output projection: [4096,4096] x [4096,3072] -> fp32 out
    {
        dim3 grid(Hdim / 128, ROWS / 128);
        gemm_bf16_wmma<0, 1><<<grid, 256, 0, stream>>>(
            attn, Wo, out, QSZ, QSZ, Hdim, Hdim);
    }
    (void)in_sizes; (void)n_in; (void)out_size; (void)ws_size;
}
